// FullPageHTRDecoder_75058848465482
// MI455X (gfx1250) — matrix operand order repaired and checked by reading, not compiled, not run on hardware
//
#include <hip/hip_runtime.h>
#include <hip/hip_bf16.h>
#include <math.h>

// FullPageHTRDecoder on MI455X (gfx1250, wave32):
//  - bf16 WMMA (v_wmma_f32_16x16x32_bf16) for every GEMM, f32 accumulate
//  - weights converted to bf16 once per call (~50MB, fits in 192MB L2)
//  - TDM tensor_load_to_lds stages the shared A-tile in LDS for the big
//    prologue K/V-projection GEMM (TENSORcnt + s_wait_tensorcnt)
//  - VALU+LDS kernels for softmax attention / layernorm / classifier
// Dims: D=512 L=6 H=8 HD=64 DFF=2048 V=80 T=64 B=8 S=576

typedef __bf16 bf16_t;
typedef __attribute__((ext_vector_type(16))) __bf16 v16bf;
typedef __attribute__((ext_vector_type(8)))  __bf16 v8bf;
typedef __attribute__((ext_vector_type(8)))  float  v8f;
typedef __attribute__((ext_vector_type(4)))  unsigned int u32x4;
typedef __attribute__((ext_vector_type(8)))  int i32x8;
typedef __attribute__((ext_vector_type(4)))  int i32x4;

#if defined(__has_builtin)
#if __has_builtin(__builtin_amdgcn_tensor_load_to_lds) && __has_builtin(__builtin_amdgcn_s_wait_tensorcnt)
#define HAVE_TDM 1
#endif
#endif
#ifndef HAVE_TDM
#define HAVE_TDM 0
#endif

// ---------------------------------------------------------------------------
// WMMA fragment loaders (layouts per CDNA5 ISA §7.12.2, wave32)
// ---------------------------------------------------------------------------

// A: 16x32 bf16, row-major source, lane holds row (lane&15).
// lanes 0-15: K = k0+{0..7, 16..23}; lanes 16-31: K = k0+{8..15, 24..31}.
__device__ __forceinline__ v16bf load_A16x32(const bf16_t* __restrict__ A, int lda, int k0) {
  const int lane = threadIdx.x & 31;
  const int row  = lane & 15;
  const int kb   = k0 + ((lane >> 4) << 3);
  const bf16_t* p = A + (size_t)row * lda + kb;
  v8bf lo = *(const v8bf*)(p);
  v8bf hi = *(const v8bf*)(p + 16);
  v16bf r;
#pragma unroll
  for (int i = 0; i < 8; ++i) { r[i] = lo[i]; r[i + 8] = hi[i]; }
  return r;
}

// B: 32x16 bf16 where B[k][n] = W[n][k], W row-major (N x K).
// lane holds column n0+(lane&15); lanes 0-15: K = k0+{0..7, 16..23}; lanes 16-31: K = k0+{8..15, 24..31}.
__device__ __forceinline__ v16bf load_Bt32x16(const bf16_t* __restrict__ W, int ldw, int n0, int k0) {
  const int lane = threadIdx.x & 31;
  const int col  = n0 + (lane & 15);
  const int kb   = k0 + ((lane >> 4) << 3);
  const bf16_t* p = W + (size_t)col * ldw + kb;
  v8bf lo = *(const v8bf*)(p);
  v8bf hi = *(const v8bf*)(p + 16);
  v16bf r;
#pragma unroll
  for (int i = 0; i < 8; ++i) { r[i] = lo[i]; r[i + 8] = hi[i]; }
  return r;
}

// 16x16 tile of C = A(16xK) * W(NxK)^T, accumulated over K in steps of 32.
__device__ __forceinline__ v8f gemm_tile(const bf16_t* __restrict__ A,
                                         const bf16_t* __restrict__ W,
                                         int n0, int K) {
  v8f c = {0.f, 0.f, 0.f, 0.f, 0.f, 0.f, 0.f, 0.f};
  for (int k0 = 0; k0 < K; k0 += 32) {
    v16bf a = load_A16x32(A, K, k0);
    v16bf b = load_Bt32x16(W, K, n0, k0);
    c = __builtin_amdgcn_wmma_f32_16x16x32_bf16(false, a, false, b, (short)0, c, false, false);
  }
  return c;
}

// C/D layout: VGPR r <-> row m = r + 8*(lane>=16); col n = n0 + (lane&15).

// ---------------------------------------------------------------------------
// Prologue kernels
// ---------------------------------------------------------------------------

__global__ void k_cast_bf16(const float* __restrict__ s, bf16_t* __restrict__ d, size_t n) {
  size_t i  = (size_t)blockIdx.x * blockDim.x + threadIdx.x;
  size_t st = (size_t)gridDim.x * blockDim.x;
  for (; i < n; i += st) d[i] = (bf16_t)s[i];
}

__global__ void k_prologue(int* __restrict__ tok, float* __restrict__ ids,
                           bf16_t* __restrict__ ctxB) {
  int tid = threadIdx.x;
  if (tid < 8) { tok[tid] = 1; ids[tid * 65] = 1.0f; }  // SOS
  for (int i = tid; i < 16 * 512; i += blockDim.x) ctxB[i] = (bf16_t)0.f;
}

// mem_k / mem_v = memory @ ca_w[l, D:2D / 2D:3D].T + bias   (M=4608, N=512, K=512)
// grid: x = 288 mtiles * 2 nblocks(256 cols) = 576, y = 12 (layer*2 + kv); block = 128.
// The 16x512 bf16 A-tile is staged in LDS once per workgroup via the Tensor Data
// Mover (async DMA, TENSORcnt) and shared by all 4 waves; each wave then computes
// a 16x64 strip (4 WMMA accumulators reusing one A fragment per k-step).
__global__ void k_memkv(const bf16_t* __restrict__ memB, const bf16_t* __restrict__ caW,
                        const float* __restrict__ ca_b,
                        bf16_t* __restrict__ memK, bf16_t* __restrict__ memV) {
  __shared__ __align__(16) bf16_t Atile[16 * 512];
  const int ly = blockIdx.y, l = ly >> 1, kv = ly & 1;
  const int mt = blockIdx.x >> 1, nb = blockIdx.x & 1;
  const int wave = threadIdx.x >> 5;
  const int m0 = mt * 16;
  const int n0 = nb * 256 + wave * 64;
  const bf16_t* Ag = memB + (size_t)m0 * 512;

#if HAVE_TDM
  if (threadIdx.x < 32) {
    // Build Tensor DMA Descriptor (D#) per CDNA5 ISA:
    // 1-D tile of 8192 bf16 elements (16 rows x 512, contiguous), global -> LDS.
    unsigned long long ga = (unsigned long long)(uintptr_t)Ag;
    unsigned int lds = (unsigned int)(uintptr_t)Atile;  // LDS offset = addr[31:0]
    u32x4 g0 = { 1u,                                            // count=1 (valid user D#)
                 lds,                                           // lds_addr
                 (unsigned int)(ga & 0xffffffffu),              // global_addr[31:0]
                 (unsigned int)(((ga >> 32) & 0x1ffffffull)     // global_addr[56:32]
                                | 0x80000000u) };               // type=2 ("image")
    i32x8 g1 = { 0x00010000,   // workgroup_mask=0 (not in cluster), data_size=1 -> 2B
                 0x20000000,   // [31:16] tensor_dim0[15:0] = 8192
                 0x00010000,   // [15:0] tensor_dim0[31:16]=0, [31:16] tensor_dim1=1
                 0x20000000,   // [15:0] tensor_dim1[31:16]=0, [31:16] tile_dim0 = 8192
                 0,            // tile_dim1=0, tile_dim2=0
                 8192,         // tensor_dim0_stride[31:0]
                 0, 0 };
    i32x4 gz  = { 0, 0, 0, 0 };
    i32x8 gz8 = { 0, 0, 0, 0, 0, 0, 0, 0 };
    // 6-arg form (clang-23 / therock-10.0 toolchain): (g0, g1, g2, g3, g4, cpol)
    __builtin_amdgcn_tensor_load_to_lds(g0, g1, gz, gz, gz8, 0);
    __builtin_amdgcn_s_wait_tensorcnt(0);  // issuing wave waits for TDM completion
  }
  __syncthreads();                          // publish LDS tile to the other waves
#else
  for (int i = threadIdx.x * 8; i < 16 * 512; i += blockDim.x * 8)
    *(v8bf*)(Atile + i) = *(const v8bf*)(Ag + i);
  __syncthreads();
#endif

  const bf16_t* Wp = caW + ((size_t)l * 1536 + (kv ? 1024 : 512)) * 512;
  v8f c[4];
#pragma unroll
  for (int i = 0; i < 4; ++i) c[i] = (v8f){0.f, 0.f, 0.f, 0.f, 0.f, 0.f, 0.f, 0.f};
  for (int k0 = 0; k0 < 512; k0 += 32) {
    v16bf a = load_A16x32(Atile, 512, k0);     // ds_load_b128 path
#pragma unroll
    for (int i = 0; i < 4; ++i) {
      v16bf b = load_Bt32x16(Wp, 512, n0 + i * 16, k0);
      c[i] = __builtin_amdgcn_wmma_f32_16x16x32_bf16(false, a, false, b, (short)0, c[i],
                                                     false, false);
    }
  }

  const float* bias = ca_b + l * 1536 + (kv ? 1024 : 512);
  bf16_t* outp = (kv ? memV : memK) + (size_t)l * 4608 * 512;
  const int lane = threadIdx.x & 31;
  const int mb = (lane >> 4) * 8;
#pragma unroll
  for (int i = 0; i < 4; ++i) {
    int n = n0 + i * 16 + (lane & 15);
#pragma unroll
    for (int r = 0; r < 8; ++r) {
      int m = m0 + mb + r;
      outp[(size_t)m * 512 + n] = (bf16_t)(c[i][r] + bias[n]);
    }
  }
}

// ---------------------------------------------------------------------------
// Per-step kernels
// ---------------------------------------------------------------------------

// x = emb[tok]*sqrt(D) + pe[pos]; pad rows 8..15 of xB with zeros. grid 16, block 128.
__global__ void k_embed(const int* __restrict__ tok, const float* __restrict__ emb,
                        float* __restrict__ xF, bf16_t* __restrict__ xB, int t) {
  const int m = blockIdx.x;
  const int pos = (t == 0) ? 0 : (t + 1);
  for (int d = threadIdx.x; d < 512; d += blockDim.x) {
    float val = 0.f;
    if (m < 8) {
      int i2 = d & ~1;
      float ang = (float)pos * expf(-(float)i2 * (9.21034037197618f / 512.f)); // ln(10000)/D
      float pe  = (d & 1) ? cosf(ang) : sinf(ang);
      val = emb[(size_t)tok[m] * 512 + d] * 22.62741699796952f + pe;          // sqrt(512)
      xF[m * 512 + d] = val;
    }
    xB[m * 512 + d] = (bf16_t)val;
  }
}

// qkv = x @ sa_w[l].T + sa_b[l]; scatter q->qbuf, k/v -> caches at time t.
// grid 24 (N=1536), block 128.
__global__ void k_qkv(const bf16_t* __restrict__ A, const bf16_t* __restrict__ Wp,
                      const float* __restrict__ bias, float* __restrict__ qout,
                      float* __restrict__ kcache, float* __restrict__ vcache,
                      int l, int t) {
  const int wave = threadIdx.x >> 5;
  const int n0 = blockIdx.x * 64 + wave * 16;
  __builtin_prefetch(Wp + (size_t)(n0 + (threadIdx.x & 15)) * 512, 0, 0); // global_prefetch_b8
  v8f c = gemm_tile(A, Wp, n0, 512);
  const int lane = threadIdx.x & 31;
  const int n  = n0 + (lane & 15);
  const int mb = (lane >> 4) * 8;
#pragma unroll
  for (int r = 0; r < 8; ++r) {
    int m = mb + r;
    if (m < 8) {
      float val = c[r] + bias[n];
      if (n < 512)       qout[m * 512 + n] = val;
      else if (n < 1024) kcache[((size_t)(l * 8 + m) * 64 + t) * 512 + (n - 512)]  = val;
      else               vcache[((size_t)(l * 8 + m) * 64 + t) * 512 + (n - 1024)] = val;
    }
  }
}

// out = A @ W^T + bias + resid (M=8 valid rows, N=512). grid 8, block 128.
__global__ void k_proj_resid(const bf16_t* __restrict__ A, const bf16_t* __restrict__ Wp,
                             const float* __restrict__ bias, const float* __restrict__ resid,
                             float* __restrict__ outF, int K) {
  const int wave = threadIdx.x >> 5;
  const int n0 = blockIdx.x * 64 + wave * 16;
  __builtin_prefetch(Wp + (size_t)(n0 + (threadIdx.x & 15)) * K, 0, 0);
  v8f c = gemm_tile(A, Wp, n0, K);
  const int lane = threadIdx.x & 31;
  const int n  = n0 + (lane & 15);
  const int mb = (lane >> 4) * 8;
#pragma unroll
  for (int r = 0; r < 8; ++r) {
    int m = mb + r;
    if (m < 8) outF[m * 512 + n] = c[r] + bias[n] + resid[m * 512 + n];
  }
}

// q = x @ ca_w[l,:D].T + ca_b[l,:D]. grid 8, block 128.
__global__ void k_qproj(const bf16_t* __restrict__ A, const bf16_t* __restrict__ Wp,
                        const float* __restrict__ bias, float* __restrict__ qout) {
  const int wave = threadIdx.x >> 5;
  const int n0 = blockIdx.x * 64 + wave * 16;
  __builtin_prefetch(Wp + (size_t)(n0 + (threadIdx.x & 15)) * 512, 0, 0);
  v8f c = gemm_tile(A, Wp, n0, 512);
  const int lane = threadIdx.x & 31;
  const int n  = n0 + (lane & 15);
  const int mb = (lane >> 4) * 8;
#pragma unroll
  for (int r = 0; r < 8; ++r) {
    int m = mb + r;
    if (m < 8) qout[m * 512 + n] = c[r] + bias[n];
  }
}

// h = gelu(x @ ff1_w.T + ff1_b) -> bf16, pad rows zeroed. grid 32 (N=2048), block 128.
__global__ void k_ff1(const bf16_t* __restrict__ A, const bf16_t* __restrict__ Wp,
                      const float* __restrict__ bias, bf16_t* __restrict__ outB) {
  const int wave = threadIdx.x >> 5;
  const int n0 = blockIdx.x * 64 + wave * 16;
  __builtin_prefetch(Wp + (size_t)(n0 + (threadIdx.x & 15)) * 512, 0, 0);
  v8f c = gemm_tile(A, Wp, n0, 512);
  const int lane = threadIdx.x & 31;
  const int n  = n0 + (lane & 15);
  const int mb = (lane >> 4) * 8;
#pragma unroll
  for (int r = 0; r < 8; ++r) {
    int m = mb + r;
    float val = 0.f;
    if (m < 8) {
      float x = c[r] + bias[n];
      val = 0.5f * x * (1.f + erff(x * 0.7071067811865475f));  // exact gelu
    }
    outB[(size_t)m * 2048 + n] = (bf16_t)val;                  // rows 8..15 -> 0
  }
}

// LayerNorm of one row (512). grid 8, block 128.
__global__ void k_ln(const float* __restrict__ y, const float* __restrict__ g,
                     const float* __restrict__ bta, float* __restrict__ xF,
                     bf16_t* __restrict__ xB) {
  const int row = blockIdx.x, tid = threadIdx.x;
  __shared__ float sred[128];
  const float* yr = y + row * 512;
  float s = 0.f;
  for (int d = tid; d < 512; d += 128) s += yr[d];
  sred[tid] = s; __syncthreads();
  for (int o = 64; o > 0; o >>= 1) { if (tid < o) sred[tid] += sred[tid + o]; __syncthreads(); }
  float mean = sred[0] * (1.f / 512.f); __syncthreads();
  float v = 0.f;
  for (int d = tid; d < 512; d += 128) { float t2 = yr[d] - mean; v += t2 * t2; }
  sred[tid] = v; __syncthreads();
  for (int o = 64; o > 0; o >>= 1) { if (tid < o) sred[tid] += sred[tid + o]; __syncthreads(); }
  float rstd = rsqrtf(sred[0] * (1.f / 512.f) + 1e-5f);
  for (int d = tid; d < 512; d += 128) {
    float val = (yr[d] - mean) * rstd * g[d] + bta[d];
    xF[row * 512 + d] = val;
    xB[row * 512 + d] = (bf16_t)val;
  }
}

// Causal self-attention over KV cache (<= t+1 keys). grid 64 (b,h), block 64.
__global__ void k_self_attn(const float* __restrict__ q, const float* __restrict__ kc,
                            const float* __restrict__ vc, float* __restrict__ ctxF,
                            bf16_t* __restrict__ ctxB, int l, int t) {
  const int b = blockIdx.x >> 3, h = blockIdx.x & 7;
  const int tid = threadIdx.x;
  __shared__ float p[64];
  __shared__ float red[64];
  const float* qp = q + b * 512 + h * 64;
  const float* kb = kc + ((size_t)(l * 8 + b) * 64) * 512 + h * 64;
  float sc = -1e30f;
  if (tid <= t) {
    const float* kp = kb + (size_t)tid * 512;
    float acc = 0.f;
#pragma unroll 8
    for (int e = 0; e < 64; ++e) acc += qp[e] * kp[e];
    sc = acc * 0.125f;  // 1/sqrt(HD)
  }
  red[tid] = sc; __syncthreads();
  for (int o = 32; o > 0; o >>= 1) { if (tid < o) red[tid] = fmaxf(red[tid], red[tid + o]); __syncthreads(); }
  float mx = red[0]; __syncthreads();
  float e1 = (tid <= t) ? __expf(sc - mx) : 0.f;
  p[tid] = e1;
  red[tid] = e1; __syncthreads();
  for (int o = 32; o > 0; o >>= 1) { if (tid < o) red[tid] += red[tid + o]; __syncthreads(); }
  float rs = 1.f / red[0];
  const float* vb = vc + ((size_t)(l * 8 + b) * 64) * 512 + h * 64;
  float acc = 0.f;
  for (int i = 0; i <= t; ++i) acc += p[i] * vb[(size_t)i * 512 + tid];
  acc *= rs;
  ctxF[b * 512 + h * 64 + tid] = acc;
  ctxB[b * 512 + h * 64 + tid] = (bf16_t)acc;
}

// Cross-attention over S=576 memory keys (bf16 K/V). grid 64 (b,h), block 64.
__global__ void k_cross_attn(const float* __restrict__ q, const bf16_t* __restrict__ memK,
                             const bf16_t* __restrict__ memV, float* __restrict__ ctxF,
                             bf16_t* __restrict__ ctxB, int l) {
  const int b = blockIdx.x >> 3, h = blockIdx.x & 7;
  const int tid = threadIdx.x;
  __shared__ float p[576];
  __shared__ float red[64];
  const float* qp = q + b * 512 + h * 64;
  const bf16_t* kb = memK + ((size_t)l * 4608 + b * 576) * 512 + h * 64;
  float lmax = -1e30f;
  for (int s = tid; s < 576; s += 64) {
    const bf16_t* kp = kb + (size_t)s * 512;
    float acc = 0.f;
#pragma unroll 8
    for (int e = 0; e < 64; ++e) acc += qp[e] * (float)kp[e];
    acc *= 0.125f;
    p[s] = acc;
    lmax = fmaxf(lmax, acc);
  }
  red[tid] = lmax; __syncthreads();
  for (int o = 32; o > 0; o >>= 1) { if (tid < o) red[tid] = fmaxf(red[tid], red[tid + o]); __syncthreads(); }
  float mx = red[0]; __syncthreads();
  float lsum = 0.f;
  for (int s = tid; s < 576; s += 64) { float e2 = __expf(p[s] - mx); p[s] = e2; lsum += e2; }
  red[tid] = lsum; __syncthreads();
  for (int o = 32; o > 0; o >>= 1) { if (tid < o) red[tid] += red[tid + o]; __syncthreads(); }
  float rs = 1.f / red[0];
  const bf16_t* vb = memV + ((size_t)l * 4608 + b * 576) * 512 + h * 64;
  float acc = 0.f;
  for (int s = 0; s < 576; ++s) acc += p[s] * (float)vb[(size_t)s * 512 + tid];
  acc *= rs;
  ctxF[b * 512 + h * 64 + tid] = acc;
  ctxB[b * 512 + h * 64 + tid] = (bf16_t)acc;
}

// logits = x @ clf_w.T + clf_b ; argmax -> next token + sampled id. grid 1, block 256.
__global__ void k_logits(const float* __restrict__ x, const float* __restrict__ cw,
                         const float* __restrict__ cb, float* __restrict__ outLogits,
                         float* __restrict__ outIds, int* __restrict__ tok, int t) {
  __shared__ float lg[640];
  const int tid = threadIdx.x;
  for (int idx = tid; idx < 640; idx += blockDim.x) {
    int b = idx / 80, v = idx - b * 80;
    const float* xr = x + b * 512;
    const float* wr = cw + (size_t)v * 512;
    float acc = cb[v];
#pragma unroll 8
    for (int d = 0; d < 512; ++d) acc += xr[d] * wr[d];
    lg[idx] = acc;
    outLogits[((size_t)b * 64 + t) * 80 + v] = acc;   // (B,T,V) layout
  }
  __syncthreads();
  if (tid < 8) {
    int best = 0; float bv = lg[tid * 80];
    for (int v = 1; v < 80; ++v) { float qv = lg[tid * 80 + v]; if (qv > bv) { bv = qv; best = v; } }
    tok[tid] = best;
    outIds[tid * 65 + (t + 1)] = (float)best;
  }
}

// First-EOS masking: positions after first EOS(=2) -> PAD(=0); no EOS -> untouched.
__global__ void k_eos(float* __restrict__ ids) {
  const int b = threadIdx.x;
  if (b < 8) {
    int eos = 0;
    for (int i = 0; i < 65; ++i) { if (ids[b * 65 + i] == 2.0f) { eos = i; break; } }
    if (eos != 0) for (int i = eos + 1; i < 65; ++i) ids[b * 65 + i] = 0.f;
  }
}

// ---------------------------------------------------------------------------
// Host orchestration (graph-capture safe: only async launches on `stream`)
// ---------------------------------------------------------------------------

extern "C" void kernel_launch(void* const* d_in, const int* in_sizes, int n_in,
                              void* d_out, int out_size, void* d_ws, size_t ws_size,
                              hipStream_t stream) {
  (void)in_sizes; (void)n_in; (void)out_size; (void)ws_size;

  const float* memory = (const float*)d_in[0];
  const float* emb    = (const float*)d_in[1];
  const float* sa_w   = (const float*)d_in[2];
  const float* sa_b   = (const float*)d_in[3];
  const float* sa_ow  = (const float*)d_in[4];
  const float* sa_ob  = (const float*)d_in[5];
  const float* ca_w   = (const float*)d_in[6];
  const float* ca_b   = (const float*)d_in[7];
  const float* ca_ow  = (const float*)d_in[8];
  const float* ca_ob  = (const float*)d_in[9];
  const float* ln1_g  = (const float*)d_in[10];
  const float* ln1_b  = (const float*)d_in[11];
  const float* ln2_g  = (const float*)d_in[12];
  const float* ln2_b  = (const float*)d_in[13];
  const float* ln3_g  = (const float*)d_in[14];
  const float* ln3_b  = (const float*)d_in[15];
  const float* ff1_w  = (const float*)d_in[16];
  const float* ff1_b  = (const float*)d_in[17];
  const float* ff2_w  = (const float*)d_in[18];
  const float* ff2_b  = (const float*)d_in[19];
  const float* clf_w  = (const float*)d_in[20];
  const float* clf_b  = (const float*)d_in[21];

  float* outLogits = (float*)d_out;                 // 8*64*80
  float* outIds    = (float*)d_out + 8 * 64 * 80;   // 8*65

  // Workspace carve-out (~125 MB), 256B-aligned slots.
  char* ws = (char*)d_ws;
  size_t off = 0;
  auto alloc = [&](size_t bytes) -> char* {
    char* p = ws + off;
    off = (off + bytes + 255) & ~(size_t)255;
    return p;
  };
  bf16_t* memB = (bf16_t*)alloc(4608ull * 512 * 2);        // memory bf16
  bf16_t* saW  = (bf16_t*)alloc(6ull * 1536 * 512 * 2);
  bf16_t* saOW = (bf16_t*)alloc(6ull * 512 * 512 * 2);
  bf16_t* caW  = (bf16_t*)alloc(6ull * 1536 * 512 * 2);
  bf16_t* caOW = (bf16_t*)alloc(6ull * 512 * 512 * 2);
  bf16_t* f1W  = (bf16_t*)alloc(6ull * 2048 * 512 * 2);
  bf16_t* f2W  = (bf16_t*)alloc(6ull * 512 * 2048 * 2);
  bf16_t* memK = (bf16_t*)alloc(6ull * 4608 * 512 * 2);    // per-layer cross K
  bf16_t* memV = (bf16_t*)alloc(6ull * 4608 * 512 * 2);    // per-layer cross V
  float*  kc   = (float*)alloc(6ull * 8 * 64 * 512 * 4);   // self-attn K cache
  float*  vc   = (float*)alloc(6ull * 8 * 64 * 512 * 4);   // self-attn V cache
  float*  xF   = (float*)alloc(8ull * 512 * 4);
  bf16_t* xB   = (bf16_t*)alloc(16ull * 512 * 2);          // rows 8..15 zero-padded
  float*  qbuf = (float*)alloc(8ull * 512 * 4);
  float*  ctxF = (float*)alloc(8ull * 512 * 4);
  bf16_t* ctxB = (bf16_t*)alloc(16ull * 512 * 2);
  float*  yF   = (float*)alloc(8ull * 512 * 4);
  bf16_t* hB   = (bf16_t*)alloc(16ull * 2048 * 2);
  int*    tok  = (int*)alloc(8ull * 4);

  // --- prologue: init state, cast weights to bf16, precompute cross K/V ---
  k_prologue<<<1, 256, 0, stream>>>(tok, outIds, ctxB);
  k_cast_bf16<<<1024, 256, 0, stream>>>(memory, memB, 4608ull * 512);
  k_cast_bf16<<<1024, 256, 0, stream>>>(sa_w,  saW,  6ull * 1536 * 512);
  k_cast_bf16<<<1024, 256, 0, stream>>>(sa_ow, saOW, 6ull * 512 * 512);
  k_cast_bf16<<<1024, 256, 0, stream>>>(ca_w,  caW,  6ull * 1536 * 512);
  k_cast_bf16<<<1024, 256, 0, stream>>>(ca_ow, caOW, 6ull * 512 * 512);
  k_cast_bf16<<<1024, 256, 0, stream>>>(ff1_w, f1W,  6ull * 2048 * 512);
  k_cast_bf16<<<1024, 256, 0, stream>>>(ff2_w, f2W,  6ull * 512 * 2048);
  k_memkv<<<dim3(576, 12), 128, 0, stream>>>(memB, caW, ca_b, memK, memV);

  // --- 64 sequential decode steps ---
  for (int t = 0; t < 64; ++t) {
    k_embed<<<16, 128, 0, stream>>>(tok, emb, xF, xB, t);
    for (int l = 0; l < 6; ++l) {
      const bf16_t* saW_l  = saW  + (size_t)l * 1536 * 512;
      const bf16_t* saOW_l = saOW + (size_t)l * 512 * 512;
      const bf16_t* caW_l  = caW  + (size_t)l * 1536 * 512;   // rows 0..511 = q proj
      const bf16_t* caOW_l = caOW + (size_t)l * 512 * 512;
      const bf16_t* f1W_l  = f1W  + (size_t)l * 2048 * 512;
      const bf16_t* f2W_l  = f2W  + (size_t)l * 512 * 2048;

      k_qkv<<<24, 128, 0, stream>>>(xB, saW_l, sa_b + l * 1536, qbuf, kc, vc, l, t);
      k_self_attn<<<64, 64, 0, stream>>>(qbuf, kc, vc, ctxF, ctxB, l, t);
      k_proj_resid<<<8, 128, 0, stream>>>(ctxB, saOW_l, sa_ob + l * 512, xF, yF, 512);
      k_ln<<<8, 128, 0, stream>>>(yF, ln1_g + l * 512, ln1_b + l * 512, xF, xB);

      k_qproj<<<8, 128, 0, stream>>>(xB, caW_l, ca_b + l * 1536, qbuf);
      k_cross_attn<<<64, 64, 0, stream>>>(qbuf, memK, memV, ctxF, ctxB, l);
      k_proj_resid<<<8, 128, 0, stream>>>(ctxB, caOW_l, ca_ob + l * 512, xF, yF, 512);
      k_ln<<<8, 128, 0, stream>>>(yF, ln2_g + l * 512, ln2_b + l * 512, xF, xB);

      k_ff1<<<32, 128, 0, stream>>>(xB, f1W_l, ff1_b + l * 2048, hB);
      k_proj_resid<<<8, 128, 0, stream>>>(hB, f2W_l, ff2_b + l * 512, xF, yF, 2048);
      k_ln<<<8, 128, 0, stream>>>(yF, ln3_g + l * 512, ln3_b + l * 512, xF, xB);
    }
    k_logits<<<1, 256, 0, stream>>>(xF, clf_w, clf_b, outLogits, outIds, tok, t);
  }

  // --- epilogue: EOS -> PAD masking of sampled ids ---
  k_eos<<<1, 32, 0, stream>>>(outIds);
}
